// DNRI_52201032515959
// MI455X (gfx1250) — matrix operand order repaired and checked by reading, not compiled, not run on hardware
//
#include <hip/hip_runtime.h>
#include <hip/hip_bf16.h>

// ---------------------------------------------------------------------------
// DNRI forward step for MI455X (gfx1250, wave32, WMMA).
// Heavy path: per-edge 2-layer MLP => f16 WMMA 16x16x32 with f32 accum.
// Weights pre-packed into ISA B-fragment layout so GEMM inner loop is
// ds_load_b128 (A) + global_load_b128 (B, L2-resident) + v_wmma.
// ---------------------------------------------------------------------------

typedef __attribute__((ext_vector_type(16))) _Float16 v16h;
typedef __attribute__((ext_vector_type(8)))  _Float16 v8h;
typedef __attribute__((ext_vector_type(8)))  float    v8f;

#define BB   32
#define NV   100
#define DI   4
#define HH   256
#define KTY  4
#define EE   (NV*(NV-1))      // 9900
#define ROWS (BB*EE)          // 316800

#define A0S  520              // LDS row stride (halves) for 512-wide A, +8 pad
#define M1S  264              // LDS row stride (halves) for 256-wide A, +8 pad

// ---- fragment loaders -----------------------------------------------------

// A-matrix 16x32 f16 fragment from LDS (ISA 7.12.2 layout):
// lanes 0-15 (M=lane): K = j (j<8), K = j+8 (j>=8)      -> chunks [0..8),[16..24)
// lanes 16-31 (M=lane-16): K = j+8 (j<8), K = j+16 (>=8)-> chunks [8..16),[24..32)
__device__ __forceinline__ v16h lds_a_frag(const _Float16* __restrict__ row, int kt) {
    const int grp = (threadIdx.x >> 4) & 1;
    const v8h lo = *(const v8h*)(row + kt * 32 + grp * 8);
    const v8h hi = *(const v8h*)(row + kt * 32 + 16 + grp * 8);
    return __builtin_shufflevector(lo, hi, 0, 1, 2, 3, 4, 5, 6, 7,
                                           8, 9, 10, 11, 12, 13, 14, 15);
}

// B-matrix 32x16 f16 fragment from pre-packed global weights.
// Packed layout: frag(kt,nt) -> [32 lanes][16 halves] contiguous (32B/lane).
__device__ __forceinline__ v16h gbl_b_frag(const _Float16* __restrict__ Wp, int kt, int nt) {
    const int lane = threadIdx.x & 31;
    return *(const v16h*)(Wp + ((size_t)((kt * 16 + nt) * 32 + lane)) * 16);
}

__device__ __forceinline__ v8f bcast8(float x) {
    v8f r = { x, x, x, x, x, x, x, x };
    return r;
}
__device__ __forceinline__ float sigm(float x) { return 1.0f / (1.0f + expf(-x)); }

// ---- weight packing -------------------------------------------------------
// dst index = ti*KT16*16*512 + (kt*16+nt)*512 + lane*16 + j
// source element: k = kt*32 + (lane>=16)*8 + (j&7) + (j>=8)*16 ; n = nt*16 + (lane&15)

__global__ void pack_w1_kernel(const float* __restrict__ W1, _Float16* __restrict__ dst) {
    int idx = blockIdx.x * blockDim.x + threadIdx.x;
    if (idx >= 3 * 512 * 256) return;
    int ti = idx / (512 * 256);
    int r  = idx - ti * (512 * 256);
    int u = r & 511, lane = u >> 4, j = u & 15;
    int frag = r >> 9, kt = frag >> 4, nt = frag & 15;
    int k = kt * 32 + ((lane >> 4) << 3) + (j & 7) + ((j >> 3) << 4);
    int n = (nt << 4) + (lane & 15);
    dst[idx] = (_Float16)W1[((size_t)(ti + 1) * 512 + k) * 256 + n];
}

__global__ void pack_w2_kernel(const float* __restrict__ W2, _Float16* __restrict__ dst) {
    int idx = blockIdx.x * blockDim.x + threadIdx.x;
    if (idx >= 3 * 256 * 256) return;
    int ti = idx / (256 * 256);
    int r  = idx - ti * (256 * 256);
    int u = r & 511, lane = u >> 4, j = u & 15;
    int frag = r >> 9, kt = frag >> 4, nt = frag & 15;
    int k = kt * 32 + ((lane >> 4) << 3) + (j & 7) + ((j >> 3) << 4);
    int n = (nt << 4) + (lane & 15);
    dst[idx] = (_Float16)W2[((size_t)(ti + 1) * 256 + k) * 256 + n];
}

__global__ void pack_sq_kernel(const float* __restrict__ W, _Float16* __restrict__ dst) {
    int idx = blockIdx.x * blockDim.x + threadIdx.x;
    if (idx >= 256 * 256) return;
    int u = idx & 511, lane = u >> 4, j = u & 15;
    int frag = idx >> 9, kt = frag >> 4, nt = frag & 15;
    int k = kt * 32 + ((lane >> 4) << 3) + (j & 7) + ((j >> 3) << 4);
    int n = (nt << 4) + (lane & 15);
    dst[idx] = (_Float16)W[(size_t)k * 256 + n];
}

__global__ void zero_kernel(float* __restrict__ p, int n) {
    int i = blockIdx.x * blockDim.x + threadIdx.x;
    if (i < n) p[i] = 0.0f;
}

// ---- edge message passing (dominant kernel) -------------------------------
// Block: 256 threads = 8 waves; tile = 32 edge rows x 256 out cols.
// wave -> (m_tile = wv&1, n_base = (wv>>1)*64), 4 n-tiles of 16 per wave.

__global__ __launch_bounds__(256) void msg_kernel(
    const float* __restrict__ hidden, const float* __restrict__ edges,
    const float* __restrict__ b1, const float* __restrict__ b2,
    const int* __restrict__ send, const int* __restrict__ recv,
    const _Float16* __restrict__ W1p, const _Float16* __restrict__ W2p,
    float* __restrict__ agg)
{
    __shared__ __align__(16) _Float16 A0[32 * A0S];   // gathered [recv||send], f16
    __shared__ __align__(16) _Float16 M1[32 * M1S];   // layer-1 activations
    __shared__ float EW[32 * 3];                      // edge weights / norm
    __shared__ int   AOFF[32];                        // agg scatter base offsets

    const int t = threadIdx.x;
    const int gbase = blockIdx.x * 32;

    {   // gather hidden rows: 8 threads per row, 64 cols each
        const int r = t >> 3, part = t & 7;
        const int g = gbase + r;
        const int b = g / EE;
        const int e = g - b * EE;
        const int rv = recv[e], sd = send[e];
        const float* hr = hidden + ((size_t)b * NV + rv) * HH;
        const float* hs = hidden + ((size_t)b * NV + sd) * HH;
        _Float16* dstA = A0 + r * A0S;
        const int c0 = part * 64;
        #pragma unroll 8
        for (int c = c0; c < c0 + 64; ++c) {
            float v = (c < HH) ? hr[c] : hs[c - HH];
            dstA[c] = (_Float16)v;
        }
        if (part == 0) {
            AOFF[r] = (b * NV + rv) * HH;
            const float invk = 1.0f / 3.0f;   // 1/(K-1), SKIP_FIRST
            const float* ep = edges + ((size_t)b * EE + e) * KTY;
            EW[r * 3 + 0] = ep[1] * invk;
            EW[r * 3 + 1] = ep[2] * invk;
            EW[r * 3 + 2] = ep[3] * invk;
        }
    }
    __syncthreads();

    const int wv = t >> 5, lane = t & 31;
    const int ln = lane & 15, grp = lane >> 4;
    const int mt = wv & 1;
    const int nb = (wv >> 1) * 64;
    const _Float16* aRow1 = A0 + (mt * 16 + ln) * A0S;
    const _Float16* aRow2 = M1 + (mt * 16 + ln) * M1S;

    v8f msg[4];
    #pragma unroll
    for (int nt = 0; nt < 4; ++nt) msg[nt] = bcast8(0.0f);

    for (int it = 0; it < 3; ++it) {
        const int ty = it + 1;
        // ---- layer 1: [32,512] x [512,256], K-tiles of 32 ----
        v8f c1[4];
        #pragma unroll
        for (int nt = 0; nt < 4; ++nt)
            c1[nt] = bcast8(b1[ty * HH + nb + nt * 16 + ln]);
        const _Float16* W1t = W1p + (size_t)it * (16 * 16 * 512);
        for (int kt = 0; kt < 16; ++kt) {
            if (kt + 1 < 16)
                __builtin_prefetch(W1t + (size_t)(kt + 1) * (16 * 512), 0, 1);
            v16h a = lds_a_frag(aRow1, kt);
            #pragma unroll
            for (int nt = 0; nt < 4; ++nt) {
                v16h bf = gbl_b_frag(W1t, kt, (nb >> 4) + nt);
                c1[nt] = __builtin_amdgcn_wmma_f32_16x16x32_f16(
                    false, a, false, bf, (short)0, c1[nt], false, false);
            }
        }
        #pragma unroll
        for (int nt = 0; nt < 4; ++nt) {
            #pragma unroll
            for (int v = 0; v < 8; ++v) {
                const int row = mt * 16 + v + 8 * grp;   // C/D layout: M = v + 8*(lane/16)
                M1[row * M1S + nb + nt * 16 + ln] = (_Float16)tanhf(c1[nt][v]);
            }
        }
        __syncthreads();

        // ---- layer 2: [32,256] x [256,256] ----
        v8f c2[4];
        #pragma unroll
        for (int nt = 0; nt < 4; ++nt)
            c2[nt] = bcast8(b2[ty * HH + nb + nt * 16 + ln]);
        const _Float16* W2t = W2p + (size_t)it * (8 * 16 * 512);
        for (int kt = 0; kt < 8; ++kt) {
            if (kt + 1 < 8)
                __builtin_prefetch(W2t + (size_t)(kt + 1) * (16 * 512), 0, 1);
            v16h a = lds_a_frag(aRow2, kt);
            #pragma unroll
            for (int nt = 0; nt < 4; ++nt) {
                v16h bf = gbl_b_frag(W2t, kt, (nb >> 4) + nt);
                c2[nt] = __builtin_amdgcn_wmma_f32_16x16x32_f16(
                    false, a, false, bf, (short)0, c2[nt], false, false);
            }
        }
        #pragma unroll
        for (int nt = 0; nt < 4; ++nt) {
            #pragma unroll
            for (int v = 0; v < 8; ++v) {
                const int rl = mt * 16 + v + 8 * grp;
                msg[nt][v] += tanhf(c2[nt][v]) * EW[rl * 3 + it];
            }
        }
        __syncthreads();   // M1 reused next edge type
    }

    // ---- scatter-sum to receivers: agg[b, recv, :] += msg / (N-1) ----
    const float invn = 1.0f / (float)(NV - 1);
    #pragma unroll
    for (int nt = 0; nt < 4; ++nt) {
        #pragma unroll
        for (int v = 0; v < 8; ++v) {
            const int rl = mt * 16 + v + 8 * grp;
            atomicAdd(agg + AOFF[rl] + nb + nt * 16 + ln, msg[nt][v] * invn);
        }
    }
}

// ---- GRU update + output MLP ----------------------------------------------
// Block: 128 threads = 4 waves; tile = 16 rows (b,n) x 256 cols.
// wave wv -> cols [wv*64, wv*64+64), 4 n-tiles of 16.

__global__ __launch_bounds__(128) void gru_out_kernel(
    const float* __restrict__ inputs, const float* __restrict__ hidden,
    const float* __restrict__ agg,
    const float* __restrict__ Wir, const float* __restrict__ bir,
    const float* __restrict__ Wii, const float* __restrict__ bii,
    const float* __restrict__ Win, const float* __restrict__ binn,
    const float* __restrict__ bo1, const float* __restrict__ bo2,
    const float* __restrict__ Wo3, const float* __restrict__ bo3,
    const _Float16* __restrict__ Whr_p, const _Float16* __restrict__ Whi_p,
    const _Float16* __restrict__ Whh_p,
    const _Float16* __restrict__ Wo1p, const _Float16* __restrict__ Wo2p,
    float* __restrict__ pred_out, float* __restrict__ hid_out)
{
    __shared__ __align__(16) _Float16 AG[16 * M1S];
    __shared__ __align__(16) _Float16 HB[16 * M1S];
    __shared__ __align__(16) _Float16 P1[16 * M1S];
    __shared__ __align__(16) _Float16 P2[16 * M1S];
    __shared__ float XI[16 * DI];

    const int t = threadIdx.x;
    const int gbase = blockIdx.x * 16;

    for (int i = t; i < 16 * HH; i += 128) {
        int r = i >> 8, c = i & 255;
        AG[r * M1S + c] = (_Float16)agg[(size_t)(gbase + r) * HH + c];
    }
    if (t < 16 * DI) XI[t] = inputs[(size_t)gbase * DI + t];
    __syncthreads();

    const int wv = t >> 5, lane = t & 31;
    const int ln = lane & 15, grp = lane >> 4;
    const int nb = wv * 64;
    const _Float16* aAgg = AG + ln * M1S;

    // agg @ {Whr, Whi, Whh}
    v8f aR[4], aI[4], aN[4];
    #pragma unroll
    for (int nt = 0; nt < 4; ++nt) { aR[nt] = bcast8(0.f); aI[nt] = bcast8(0.f); aN[nt] = bcast8(0.f); }
    for (int kt = 0; kt < 8; ++kt) {
        v16h a = lds_a_frag(aAgg, kt);
        #pragma unroll
        for (int nt = 0; nt < 4; ++nt) {
            const int ng = (nb >> 4) + nt;
            v16h br = gbl_b_frag(Whr_p, kt, ng);
            aR[nt] = __builtin_amdgcn_wmma_f32_16x16x32_f16(false, a, false, br, (short)0, aR[nt], false, false);
            v16h bi = gbl_b_frag(Whi_p, kt, ng);
            aI[nt] = __builtin_amdgcn_wmma_f32_16x16x32_f16(false, a, false, bi, (short)0, aI[nt], false, false);
            v16h bn = gbl_b_frag(Whh_p, kt, ng);
            aN[nt] = __builtin_amdgcn_wmma_f32_16x16x32_f16(false, a, false, bn, (short)0, aN[nt], false, false);
        }
    }

    // elementwise GRU (input matmuls are D=4 scalar FMAs)
    #pragma unroll
    for (int nt = 0; nt < 4; ++nt) {
        const int n = nb + nt * 16 + ln;
        const float wr0 = Wir[n], wr1 = Wir[HH + n], wr2 = Wir[2 * HH + n], wr3 = Wir[3 * HH + n];
        const float wi0 = Wii[n], wi1 = Wii[HH + n], wi2 = Wii[2 * HH + n], wi3 = Wii[3 * HH + n];
        const float wn0 = Win[n], wn1 = Win[HH + n], wn2 = Win[2 * HH + n], wn3 = Win[3 * HH + n];
        const float br_ = bir[n], bi_ = bii[n], bn_ = binn[n];
        #pragma unroll
        for (int v = 0; v < 8; ++v) {
            const int m = v + 8 * grp;
            const int g = gbase + m;
            const float x0 = XI[m * 4 + 0], x1 = XI[m * 4 + 1], x2 = XI[m * 4 + 2], x3 = XI[m * 4 + 3];
            const float xr = br_ + x0 * wr0 + x1 * wr1 + x2 * wr2 + x3 * wr3;
            const float xi = bi_ + x0 * wi0 + x1 * wi1 + x2 * wi2 + x3 * wi3;
            const float xn = bn_ + x0 * wn0 + x1 * wn1 + x2 * wn2 + x3 * wn3;
            const float r  = sigm(xr + aR[nt][v]);
            const float ig = sigm(xi + aI[nt][v]);
            const float nn = tanhf(xn + r * aN[nt][v]);
            const float ho = hidden[(size_t)g * HH + n];
            const float hn = (1.0f - ig) * nn + ig * ho;
            hid_out[(size_t)g * HH + n] = hn;
            HB[m * M1S + n] = (_Float16)hn;
        }
    }
    __syncthreads();

    // MLP layer 1: relu(h @ Wo1 + bo1)
    v8f p1[4];
    #pragma unroll
    for (int nt = 0; nt < 4; ++nt) p1[nt] = bcast8(bo1[nb + nt * 16 + ln]);
    const _Float16* aHB = HB + ln * M1S;
    for (int kt = 0; kt < 8; ++kt) {
        v16h a = lds_a_frag(aHB, kt);
        #pragma unroll
        for (int nt = 0; nt < 4; ++nt) {
            v16h bf = gbl_b_frag(Wo1p, kt, (nb >> 4) + nt);
            p1[nt] = __builtin_amdgcn_wmma_f32_16x16x32_f16(false, a, false, bf, (short)0, p1[nt], false, false);
        }
    }
    #pragma unroll
    for (int nt = 0; nt < 4; ++nt)
        #pragma unroll
        for (int v = 0; v < 8; ++v) {
            const int m = v + 8 * grp;
            P1[m * M1S + nb + nt * 16 + ln] = (_Float16)fmaxf(p1[nt][v], 0.0f);
        }
    __syncthreads();

    // MLP layer 2: relu(p1 @ Wo2 + bo2)
    v8f p2[4];
    #pragma unroll
    for (int nt = 0; nt < 4; ++nt) p2[nt] = bcast8(bo2[nb + nt * 16 + ln]);
    const _Float16* aP1 = P1 + ln * M1S;
    for (int kt = 0; kt < 8; ++kt) {
        v16h a = lds_a_frag(aP1, kt);
        #pragma unroll
        for (int nt = 0; nt < 4; ++nt) {
            v16h bf = gbl_b_frag(Wo2p, kt, (nb >> 4) + nt);
            p2[nt] = __builtin_amdgcn_wmma_f32_16x16x32_f16(false, a, false, bf, (short)0, p2[nt], false, false);
        }
    }
    #pragma unroll
    for (int nt = 0; nt < 4; ++nt)
        #pragma unroll
        for (int v = 0; v < 8; ++v) {
            const int m = v + 8 * grp;
            P2[m * M1S + nb + nt * 16 + ln] = (_Float16)fmaxf(p2[nt][v], 0.0f);
        }
    __syncthreads();

    // pred = inputs + p2 @ Wo3 + bo3  (N=4: per-element reduction)
    if (t < 16 * DI) {
        const int m = t >> 2, d = t & 3;
        float acc = bo3[d];
        for (int k2 = 0; k2 < HH; ++k2)
            acc += (float)P2[m * M1S + k2] * Wo3[k2 * DI + d];
        const int g = gbase + m;
        pred_out[(size_t)g * DI + d] = inputs[(size_t)g * DI + d] + acc;
    }
}

// ---------------------------------------------------------------------------

extern "C" void kernel_launch(void* const* d_in, const int* in_sizes, int n_in,
                              void* d_out, int out_size, void* d_ws, size_t ws_size,
                              hipStream_t stream) {
    const float* inputs = (const float*)d_in[0];
    const float* hidden = (const float*)d_in[1];
    const float* edges  = (const float*)d_in[2];
    const float* msg_W1 = (const float*)d_in[3];
    const float* msg_b1 = (const float*)d_in[4];
    const float* msg_W2 = (const float*)d_in[5];
    const float* msg_b2 = (const float*)d_in[6];
    const float* Whr    = (const float*)d_in[7];
    const float* Whi    = (const float*)d_in[8];
    const float* Whh    = (const float*)d_in[9];
    const float* Wir    = (const float*)d_in[10];
    const float* bir    = (const float*)d_in[11];
    const float* Wii    = (const float*)d_in[12];
    const float* bii    = (const float*)d_in[13];
    const float* Win    = (const float*)d_in[14];
    const float* binn   = (const float*)d_in[15];
    const float* Wo1    = (const float*)d_in[16];
    const float* bo1    = (const float*)d_in[17];
    const float* Wo2    = (const float*)d_in[18];
    const float* bo2    = (const float*)d_in[19];
    const float* Wo3    = (const float*)d_in[20];
    const float* bo3    = (const float*)d_in[21];
    const int*   send   = (const int*)d_in[22];
    const int*   recv   = (const int*)d_in[23];

    char* ws = (char*)d_ws;
    float*    agg  = (float*)ws;                       // 819200 f32 = 3,276,800 B
    _Float16* W1p  = (_Float16*)(ws + 3276800);        //   786,432 B
    _Float16* W2p  = (_Float16*)(ws + 4063232);        //   393,216 B
    _Float16* WhrP = (_Float16*)(ws + 4456448);        //   131,072 B each
    _Float16* WhiP = (_Float16*)(ws + 4587520);
    _Float16* WhhP = (_Float16*)(ws + 4718592);
    _Float16* Wo1P = (_Float16*)(ws + 4849664);
    _Float16* Wo2P = (_Float16*)(ws + 4980736);

    float* pred_out = (float*)d_out;                   // [B,N,D] = 12800
    float* hid_out  = pred_out + BB * NV * DI;         // [B,N,H] = 819200

    zero_kernel<<<(BB * NV * HH + 255) / 256, 256, 0, stream>>>(agg, BB * NV * HH);

    pack_w1_kernel<<<(3 * 512 * 256) / 256, 256, 0, stream>>>(msg_W1, W1p);
    pack_w2_kernel<<<(3 * 256 * 256) / 256, 256, 0, stream>>>(msg_W2, W2p);
    pack_sq_kernel<<<(256 * 256) / 256, 256, 0, stream>>>(Whr, WhrP);
    pack_sq_kernel<<<(256 * 256) / 256, 256, 0, stream>>>(Whi, WhiP);
    pack_sq_kernel<<<(256 * 256) / 256, 256, 0, stream>>>(Whh, WhhP);
    pack_sq_kernel<<<(256 * 256) / 256, 256, 0, stream>>>(Wo1, Wo1P);
    pack_sq_kernel<<<(256 * 256) / 256, 256, 0, stream>>>(Wo2, Wo2P);

    msg_kernel<<<ROWS / 32, 256, 0, stream>>>(hidden, edges, msg_b1, msg_b2,
                                              send, recv, W1p, W2p, agg);

    gru_out_kernel<<<(BB * NV) / 16, 128, 0, stream>>>(
        inputs, hidden, agg, Wir, bir, Wii, bii, Win, binn,
        bo1, bo2, Wo3, bo3, WhrP, WhiP, WhhP, Wo1P, Wo2P,
        pred_out, hid_out);
}
